// SparseAttention_59133109731390
// MI455X (gfx1250) — matrix operand order repaired and checked by reading, not compiled, not run on hardware
//
#include <hip/hip_runtime.h>
#include <hip/hip_bf16.h>

// ---------------------------------------------------------------------------
// Sparse attention for MI455X (gfx1250, wave32, WMMA + TDM).
// Shapes fixed per reference: B=2, S=2048, H=1024, nh=16, hd=64.
// ---------------------------------------------------------------------------

#define S_LEN   2048
#define H_DIM   1024
#define NHEADS  16
#define HD      64
#define BATCH   2
#define NTOK    (BATCH * S_LEN)                  // 4096
#define QKV_ELEMS (BATCH * NHEADS * S_LEN * HD)  // 4,194,304 per tensor
#define NKT     (S_LEN / 64)                     // 32 key tiles

typedef float  v8f   __attribute__((ext_vector_type(8)));
typedef __bf16 v16bf __attribute__((ext_vector_type(16)));
typedef unsigned int v4u __attribute__((ext_vector_type(4)));
typedef int v8i __attribute__((ext_vector_type(8)));
typedef int v4i __attribute__((ext_vector_type(4)));

union B32x16 { uint4 q[2]; v16bf v; };   // 32 bytes: one WMMA bf16 A/B operand

// float -> bf16 round-to-nearest-even
static __device__ __forceinline__ unsigned short f2bf(float x) {
    unsigned u = __float_as_uint(x);
    unsigned r = ((u >> 16) & 1u) + 0x7fffu;
    return (unsigned short)((u + r) >> 16);
}
static __device__ __forceinline__ unsigned pack2bf(float a, float b) {
    return (unsigned)f2bf(a) | ((unsigned)f2bf(b) << 16);
}

// lowbias32 integer hash for the pseudo-random mask columns (~10% density)
static __device__ __forceinline__ unsigned hash32(unsigned x) {
    x ^= x >> 16; x *= 0x7feb352du;
    x ^= x >> 15; x *= 0x846ca68bu;
    x ^= x >> 16;
    return x;
}

static __device__ __forceinline__ v8f wmma_bf16(const v16bf& a, const v16bf& b, v8f c) {
    return __builtin_amdgcn_wmma_f32_16x16x32_bf16(false, a, false, b, (short)0, c,
                                                   false, false);
}

// Issue one TDM 2-D tile load (64 x 64 bf16) global -> LDS, with LDS padding
// of 4 dwords every 32 dwords (row stride 72 bf16 = conflict-free b128 reads).
// D# layout per CDNA5 ISA 8.3/8.4; type=2 ("image"), data_size=1 (2 bytes).
// Toolchain here is the 6-arg builtin form (clang-23 / therock headers).
static __device__ __forceinline__ void tdm_load_tile64(unsigned lds_byte_addr,
                                                       unsigned long long gaddr,
                                                       unsigned tensor_dim0,
                                                       unsigned tensor_dim1,
                                                       unsigned stride0)
{
    v4u g0 = { 1u,                                   // count=1, user descriptor
               lds_byte_addr,
               (unsigned)(gaddr & 0xffffffffull),
               (unsigned)((gaddr >> 32) & 0x1ffffffull) | (2u << 30) };
    v8i g1 = { (int)((1u << 16)      // data_size = 2 bytes
                   | (1u << 20)      // pad_enable
                   | (4u << 22)      // pad_interval code 4 = every 32 dwords
                   | (3u << 25)),    // pad_amount  code 3 = 4 dwords (16B)
               (int)((tensor_dim0 & 0xffffu) << 16),
               (int)((tensor_dim0 >> 16) | ((tensor_dim1 & 0xffffu) << 16)),
               (int)((tensor_dim1 >> 16) | (64u << 16)),   // tile_dim0 = 64
               (int)64u,                                   // tile_dim1 = 64, tile_dim2 = 0
               (int)stride0,                               // tensor_dim0_stride lo32
               0, 0 };
    v4i z4 = { 0, 0, 0, 0 };
    v8i z8 = { 0, 0, 0, 0, 0, 0, 0, 0 };
    __builtin_amdgcn_tensor_load_to_lds(g0, g1, z4, z4, z8, 0);
}

// ---------------------------------------------------------------------------
// Kernel 0: sparse-mask bitfield precompute.
// maskT[col][q/32] bit q%32 = band(|q-col|<=2) OR ~10% hashed random.
// ---------------------------------------------------------------------------
__global__ __launch_bounds__(256)
void mask_kernel(unsigned* __restrict__ maskT)
{
    int idx = blockIdx.x * 256 + threadIdx.x;   // 0 .. 2048*64-1
    int col = idx >> 6;
    int wd  = idx & 63;
    unsigned bits = 0u;
    #pragma unroll 4
    for (int j = 0; j < 32; ++j) {
        int q = wd * 32 + j;
        int dd = q - col;
        bool band = (dd <= 2) && (dd >= -2);
        unsigned hx = hash32((unsigned)(q * S_LEN + col) * 2654435761u + 42u);
        bool rnd = (hx & 0xffffu) < 6554u;
        bits |= (band || rnd) ? (1u << j) : 0u;
    }
    maskT[col * 64 + wd] = bits;
}

// ---------------------------------------------------------------------------
// Kernel 1: fused QKV projection.  Y = X @ W^T + bias, downcast to bf16.
//   mode 0 (Q): row-major  [b][h][s][d], PRE-SCALED by 1/sqrt(hd)
//   mode 1 (K): TRANSPOSED [b][h][d][s]
//   mode 2 (V): row-major  [b][h][s][d]
// ---------------------------------------------------------------------------
__global__ __launch_bounds__(256)
void qkv_proj_kernel(const float* __restrict__ X,
                     const float* __restrict__ Wq, const float* __restrict__ bq,
                     const float* __restrict__ Wk, const float* __restrict__ bk,
                     const float* __restrict__ Wv, const float* __restrict__ bv,
                     unsigned short* __restrict__ Qo,
                     unsigned short* __restrict__ Kt,
                     unsigned short* __restrict__ Vo)
{
    const int mode = blockIdx.z;
    const float* W    = (mode == 0) ? Wq : (mode == 1) ? Wk : Wv;
    const float* bias = (mode == 0) ? bq : (mode == 1) ? bk : bv;
    unsigned short* out = (mode == 0) ? Qo : (mode == 1) ? Kt : Vo;
    const float oscale = (mode == 0) ? 0.125f : 1.0f;   // fold 1/sqrt(64) into Q

    const int t0 = blockIdx.x * 128;
    const int o0 = blockIdx.y * 64;

    __shared__ __align__(16) unsigned short Xs[128 * 40];
    __shared__ __align__(16) unsigned short Ws[32 * 72];

    const int tid = threadIdx.x;
    const int w   = tid >> 5;
    const int L   = tid & 31;
    const int wr  = w & 3;
    const int wc  = w >> 2;
    const int lr  = L & 15;
    const int kb  = (L < 16) ? 0 : 8;

    v8f acc[2][2];
    #pragma unroll
    for (int mt = 0; mt < 2; ++mt)
        #pragma unroll
        for (int nt = 0; nt < 2; ++nt) acc[mt][nt] = (v8f)0.0f;

    for (int k0 = 0; k0 < H_DIM; k0 += 32) {
        #pragma unroll
        for (int i = 0; i < 4; ++i) {
            int idx = tid + i * 256;
            int row = idx >> 3, c4 = idx & 7;
            float4 xv = *(const float4*)(X + (size_t)(t0 + row) * H_DIM + k0 + c4 * 4);
            uint2 p; p.x = pack2bf(xv.x, xv.y); p.y = pack2bf(xv.z, xv.w);
            *(uint2*)(&Xs[row * 40 + c4 * 4]) = p;
        }
        #pragma unroll
        for (int i = 0; i < 2; ++i) {
            int idx = tid + i * 256;
            int row = idx >> 3, c4 = idx & 7;
            float4 wv = *(const float4*)(W + (size_t)(o0 + row) * H_DIM + k0 + c4 * 4);
            Ws[(c4 * 4 + 0) * 72 + row] = f2bf(wv.x);
            Ws[(c4 * 4 + 1) * 72 + row] = f2bf(wv.y);
            Ws[(c4 * 4 + 2) * 72 + row] = f2bf(wv.z);
            Ws[(c4 * 4 + 3) * 72 + row] = f2bf(wv.w);
        }
        __syncthreads();

        B32x16 a[2];
        #pragma unroll
        for (int mt = 0; mt < 2; ++mt) {
            int r0 = wr * 32 + mt * 16 + lr;
            a[mt].q[0] = *(const uint4*)&Xs[r0 * 40 + kb];
            a[mt].q[1] = *(const uint4*)&Xs[r0 * 40 + kb + 16];
        }
        #pragma unroll
        for (int nt = 0; nt < 2; ++nt) {
            B32x16 bm;
            int cb = wc * 32 + nt * 16;
            #pragma unroll
            for (int s = 0; s < 16; ++s)
                bm.v[s] = __builtin_bit_cast(__bf16, Ws[(kb + (s < 8 ? s : s + 8)) * 72 + cb + lr]);
            #pragma unroll
            for (int mt = 0; mt < 2; ++mt)
                acc[mt][nt] = wmma_bf16(a[mt].v, bm.v, acc[mt][nt]);
        }
        __syncthreads();
    }

    const int hi8 = (L >= 16) ? 8 : 0;
    #pragma unroll
    for (int nt = 0; nt < 2; ++nt) {
        int o = o0 + wc * 32 + nt * 16 + lr;
        float bval = bias[o];
        int h = o >> 6, d = o & 63;
        #pragma unroll
        for (int mt = 0; mt < 2; ++mt) {
            int tb = t0 + wr * 32 + mt * 16 + hi8;
            int b = tb >> 11;
            int s = tb & (S_LEN - 1);
            if (mode == 1) {
                unsigned pk[4];
                #pragma unroll
                for (int j = 0; j < 4; ++j)
                    pk[j] = pack2bf(acc[mt][nt][2 * j] + bval, acc[mt][nt][2 * j + 1] + bval);
                uint4 u; u.x = pk[0]; u.y = pk[1]; u.z = pk[2]; u.w = pk[3];
                size_t addr = (((size_t)(b * NHEADS + h) * HD + d) * S_LEN) + s;
                *(uint4*)(out + addr) = u;
            } else {
                size_t base = ((size_t)(b * NHEADS + h) * S_LEN + s) * HD + d;
                #pragma unroll
                for (int r = 0; r < 8; ++r)
                    out[base + (size_t)r * HD] = f2bf((acc[mt][nt][r] + bval) * oscale);
            }
        }
    }
}

// ---------------------------------------------------------------------------
// Kernel 2: streaming attention.  K/V tiles DMA'd to LDS via TDM (double-
// buffered, shared by all 4 waves); bitfield mask; WMMA for S=QK^T and P*V.
// ---------------------------------------------------------------------------
__global__ __launch_bounds__(128)
void sparse_attn_kernel(const unsigned short* __restrict__ Q,
                        const unsigned short* __restrict__ Kt,
                        const unsigned short* __restrict__ V,
                        const unsigned* __restrict__ maskT,
                        float* __restrict__ out)
{
    const int bh = blockIdx.y;
    const int b  = bh / NHEADS;
    const int h  = bh % NHEADS;
    const int w  = threadIdx.x >> 5;
    const int L  = threadIdx.x & 31;
    const int lr = L & 15;
    const int hi8 = (L >= 16) ? 8 : 0;
    const int q0 = blockIdx.x * 64 + w * 16;

    __shared__ __align__(16) unsigned short KsBuf[2][64 * 72];  // K tile: rows = hd-dim
    __shared__ __align__(16) unsigned short VsBuf[2][64 * 72];  // V tile: rows = key
    __shared__ __align__(16) unsigned short Pbuf[4][16 * 72];
    unsigned short* P = Pbuf[w];

    const size_t kt_head = (size_t)bh * HD * S_LEN;   // Kt[b][h][.][.]
    const size_t v_head  = (size_t)bh * S_LEN * HD;   // V [b][h][.][.]

    // ---- Q A-operands (pre-scaled by 1/sqrt(hd)) ----
    B32x16 qa[2];
    {
        size_t qbase = ((size_t)bh * S_LEN + (q0 + lr)) * HD;
        #pragma unroll
        for (int ks = 0; ks < 2; ++ks) {
            int kbb = ks * 32 + ((L < 16) ? 0 : 8);
            qa[ks].q[0] = *(const uint4*)(Q + qbase + kbb);
            qa[ks].q[1] = *(const uint4*)(Q + qbase + kbb + 16);
        }
    }

    v8f ctx[4];
    #pragma unroll
    for (int n = 0; n < 4; ++n) ctx[n] = (v8f)0.0f;
    float m_run[8], l_run[8];
    #pragma unroll
    for (int r = 0; r < 8; ++r) { m_run[r] = -3.0e38f; l_run[r] = 0.0f; }

    const float LOG2E = 1.44269504088896f;

    // ---- prologue: TDM tile 0 into buffer 0 ----
    if (w == 0) {
        tdm_load_tile64((unsigned)(uintptr_t)&KsBuf[0][0],
                        (unsigned long long)(uintptr_t)(Kt + kt_head),
                        S_LEN, HD, S_LEN);
        tdm_load_tile64((unsigned)(uintptr_t)&VsBuf[0][0],
                        (unsigned long long)(uintptr_t)(V + v_head),
                        HD, S_LEN, HD);
    }

    for (int kt = 0; kt < NKT; ++kt) {
        const int cur = kt & 1;
        // issue next tile (other buffer), then wait for current tile's 2 loads
        if (w == 0) {
            if (kt + 1 < NKT) {
                tdm_load_tile64((unsigned)(uintptr_t)&KsBuf[cur ^ 1][0],
                                (unsigned long long)(uintptr_t)(Kt + kt_head + (kt + 1) * 64),
                                S_LEN, HD, S_LEN);
                tdm_load_tile64((unsigned)(uintptr_t)&VsBuf[cur ^ 1][0],
                                (unsigned long long)(uintptr_t)(V + v_head + (size_t)(kt + 1) * 64 * HD),
                                HD, S_LEN, HD);
                __builtin_amdgcn_s_wait_tensorcnt(2);
            } else {
                __builtin_amdgcn_s_wait_tensorcnt(0);
            }
        }
        __syncthreads();
        const unsigned short* Ks = &KsBuf[cur][0];
        const unsigned short* Vs = &VsBuf[cur][0];

        // ---- scores: S = Q * K^T (K tile from LDS) ----
        v8f sacc[4];
        #pragma unroll
        for (int n = 0; n < 4; ++n) sacc[n] = (v8f)0.0f;
        #pragma unroll
        for (int ks = 0; ks < 2; ++ks) {
            #pragma unroll
            for (int n = 0; n < 4; ++n) {
                B32x16 bk;
                #pragma unroll
                for (int s = 0; s < 16; ++s)
                    bk.v[s] = __builtin_bit_cast(__bf16, Ks[(ks * 32 + hi8 + (s < 8 ? s : s + 8)) * 72 + n * 16 + lr]);
                sacc[n] = wmma_bf16(qa[ks].v, bk.v, sacc[n]);
            }
        }

        // ---- multiplicative mask from precomputed bitfield ----
        const int sb = (q0 & 31) + hi8;   // bit base for this lane-half
        #pragma unroll
        for (int n = 0; n < 4; ++n) {
            int col = kt * 64 + n * 16 + lr;
            unsigned bits = maskT[col * 64 + (q0 >> 5)];
            #pragma unroll
            for (int r = 0; r < 8; ++r) {
                unsigned bit = (bits >> (sb + r)) & 1u;
                sacc[n][r] *= (float)bit;     // masked scores -> exactly 0
            }
        }

        // ---- online softmax ----
        #pragma unroll
        for (int r = 0; r < 8; ++r) {
            float tm = fmaxf(fmaxf(sacc[0][r], sacc[1][r]), fmaxf(sacc[2][r], sacc[3][r]));
            #pragma unroll
            for (int off = 1; off < 16; off <<= 1)
                tm = fmaxf(tm, __shfl_xor(tm, off, 32));
            float m_new = fmaxf(m_run[r], tm);
            float alpha = __builtin_amdgcn_exp2f((m_run[r] - m_new) * LOG2E);
            float m2 = m_new * LOG2E;
            m_run[r] = m_new;
            float ls = 0.0f;
            #pragma unroll
            for (int n = 0; n < 4; ++n) {
                float p = __builtin_amdgcn_exp2f(__builtin_fmaf(sacc[n][r], LOG2E, -m2));
                sacc[n][r] = p;
                ls += p;
            }
            #pragma unroll
            for (int off = 1; off < 16; off <<= 1)
                ls += __shfl_xor(ls, off, 32);
            l_run[r] = l_run[r] * alpha + ls;
            #pragma unroll
            for (int n = 0; n < 4; ++n) ctx[n][r] *= alpha;
        }

        // ---- P (C-layout) -> bf16 in wave-private LDS ----
        #pragma unroll
        for (int n = 0; n < 4; ++n)
            #pragma unroll
            for (int r = 0; r < 8; ++r)
                P[(r + hi8) * 72 + n * 16 + lr] = f2bf(sacc[n][r]);

        // ---- ctx += P * V (V tile from LDS) ----
        #pragma unroll
        for (int ks = 0; ks < 2; ++ks) {
            B32x16 pa;
            int kbb = ks * 32 + ((L < 16) ? 0 : 8);
            pa.q[0] = *(const uint4*)&P[lr * 72 + kbb];
            pa.q[1] = *(const uint4*)&P[lr * 72 + kbb + 16];
            #pragma unroll
            for (int n = 0; n < 4; ++n) {
                B32x16 bv;
                #pragma unroll
                for (int s = 0; s < 16; ++s)
                    bv.v[s] = __builtin_bit_cast(__bf16, Vs[(ks * 32 + hi8 + (s < 8 ? s : s + 8)) * 72 + n * 16 + lr]);
                ctx[n] = wmma_bf16(pa.v, bv.v, ctx[n]);
            }
        }
        __syncthreads();   // all reads of this buffer done before TDM reuses it
    }

    // ---- epilogue ----
    #pragma unroll
    for (int r = 0; r < 8; ++r) {
        float inv = 1.0f / l_run[r];
        int srow = q0 + r + hi8;
        size_t base = ((size_t)b * S_LEN + srow) * H_DIM + h * HD;
        #pragma unroll
        for (int n = 0; n < 4; ++n)
            out[base + n * 16 + lr] = ctx[n][r] * inv;
    }
}

// ---------------------------------------------------------------------------
extern "C" void kernel_launch(void* const* d_in, const int* in_sizes, int n_in,
                              void* d_out, int out_size, void* d_ws, size_t ws_size,
                              hipStream_t stream)
{
    const float* X  = (const float*)d_in[0];
    const float* Wq = (const float*)d_in[1];
    const float* bq = (const float*)d_in[2];
    const float* Wk = (const float*)d_in[3];
    const float* bk = (const float*)d_in[4];
    const float* Wv = (const float*)d_in[5];
    const float* bv = (const float*)d_in[6];

    unsigned short* Qw  = (unsigned short*)d_ws;
    unsigned short* Ktw = Qw + QKV_ELEMS;
    unsigned short* Vw  = Ktw + QKV_ELEMS;
    unsigned*       Mw  = (unsigned*)(Vw + QKV_ELEMS);   // 512 KB mask bitfield

    mask_kernel<<<dim3(S_LEN * 64 / 256), 256, 0, stream>>>(Mw);

    dim3 gProj(NTOK / 128, H_DIM / 64, 3);
    qkv_proj_kernel<<<gProj, 256, 0, stream>>>(X, Wq, bq, Wk, bk, Wv, bv, Qw, Ktw, Vw);

    dim3 gAttn(S_LEN / 64, BATCH * NHEADS);
    sparse_attn_kernel<<<gAttn, 128, 0, stream>>>(Qw, Ktw, Vw, Mw, (float*)d_out);
}
